// SelfAttention_73143293051480
// MI455X (gfx1250) — matrix operand order repaired and checked by reading, not compiled, not run on hardware
//
#include <hip/hip_runtime.h>
#include <hip/hip_bf16.h>

// ---------------------------------------------------------------------------
// Causal self-attention for MI455X (gfx1250, wave32, WMMA bf16).
//   B=4, T=2048, C=1024, H=16, hd=64.
// Pipeline:
//   1) qkv_gemm : qkv = x @ W^T + b   (fp32 in, bf16 Q/K/V out to workspace)
//   2) flash_attn: causal online-softmax attention, fp32 out
// ---------------------------------------------------------------------------

typedef __attribute__((ext_vector_type(16))) __bf16 v16bf;
typedef __attribute__((ext_vector_type(8)))  float  v8f;

#define BATCH   4
#define TLEN    2048
#define CDIM    1024
#define NHEAD   16
#define HDIM    64
#define LOG2E   1.4426950408889634f

union FragBF {
    v16bf v;
    uint4 q[2];
    unsigned short u[16];
};

__device__ __forceinline__ unsigned short f2bf(float f) {
    unsigned u = __float_as_uint(f);
    unsigned r = (u + 0x7fffu + ((u >> 16) & 1u)) >> 16;   // RNE
    return (unsigned short)r;
}

__device__ __forceinline__ uint4 pack8(float4 a, float4 b) {
    uint4 r;
    r.x = (unsigned)f2bf(a.x) | ((unsigned)f2bf(a.y) << 16);
    r.y = (unsigned)f2bf(a.z) | ((unsigned)f2bf(a.w) << 16);
    r.z = (unsigned)f2bf(b.x) | ((unsigned)f2bf(b.y) << 16);
    r.w = (unsigned)f2bf(b.z) | ((unsigned)f2bf(b.w) << 16);
    return r;
}

// A-matrix 16x32 bf16 fragment (ISA 7.12.2): lane m = L%16, half = L/16.
// i=0..7  -> k = 8*half + i ; i=8..15 -> k = 16 + 8*half + i-8.
__device__ __forceinline__ v16bf ld_fragA(const unsigned short* row, int half) {
    FragBF f;
    f.q[0] = *(const uint4*)(row + 8 * half);
    f.q[1] = *(const uint4*)(row + 16 + 8 * half);
    return f.v;
}

// B-matrix 32x16 bf16 fragment: lane n = L%16 owns one column; per lane
// i=0..7 -> k = 8*half + i ; i=8..15 -> k = 16 + 8*half + i-8 (matches A),
// read from row n of the K-major (transposed) tile.
__device__ __forceinline__ v16bf ld_fragB(const unsigned short* row, int half) {
    FragBF f;
    f.q[0] = *(const uint4*)(row + 8 * half);
    f.q[1] = *(const uint4*)(row + 16 + 8 * half);
    return f.v;
}

// ---------------------------------------------------------------------------
// Kernel 1: qkv = x @ W^T + b.  M=8192, N=3072, K=1024.
// 128x128 tile per 128-thread (4-wave) block; 64x64 per wave (16 WMMA accs).
// Outputs scattered to: Q[b][h][t][d], K[b][h][t][d], Vt[b][h][d][t] (bf16).
// ---------------------------------------------------------------------------
#define LDSTR 40   // padded LDS row stride (halfs): 80B = 16B aligned, bank-clean

__global__ __launch_bounds__(128, 1) void qkv_gemm(
    const float* __restrict__ x, const float* __restrict__ W,
    const float* __restrict__ bias,
    unsigned short* __restrict__ Qb, unsigned short* __restrict__ Kb,
    unsigned short* __restrict__ Vt)
{
    __shared__ __align__(16) unsigned short As[128 * LDSTR];
    __shared__ __align__(16) unsigned short Bs[128 * LDSTR];

    const int tid   = threadIdx.x;
    const int lane  = tid & 31;
    const int wave  = tid >> 5;
    const int waveM = wave >> 1;          // 2x2 wave grid
    const int waveN = wave & 1;
    const int half  = lane >> 4;
    const int l16   = lane & 15;
    const int mBlk  = blockIdx.x * 128;   // grid.x = 64
    const int nBlk  = blockIdx.y * 128;   // grid.y = 24

    v8f acc[4][4];
    for (int i = 0; i < 4; ++i)
        for (int j = 0; j < 4; ++j) acc[i][j] = 0.0f;

    for (int kk = 0; kk < CDIM; kk += 32) {
        __syncthreads();
        // Stage A (x) and B (rows of W, i.e. columns of W^T), fp32 -> bf16.
        for (int c = 0; c < 4; ++c) {
            int idx  = tid + 128 * c;         // 512 chunks of 8 halfs
            int row  = idx >> 2;
            int col8 = (idx & 3) * 8;
            const float4* ap = (const float4*)(x + (size_t)(mBlk + row) * CDIM + kk + col8);
            *(uint4*)&As[row * LDSTR + col8] = pack8(ap[0], ap[1]);
            const float4* bp = (const float4*)(W + (size_t)(nBlk + row) * CDIM + kk + col8);
            *(uint4*)&Bs[row * LDSTR + col8] = pack8(bp[0], bp[1]);
        }
        __syncthreads();

        v16bf af[4], bf[4];
        for (int fr = 0; fr < 4; ++fr)
            af[fr] = ld_fragA(&As[(waveM * 64 + fr * 16 + l16) * LDSTR], half);
        for (int fn = 0; fn < 4; ++fn)
            bf[fn] = ld_fragB(&Bs[(waveN * 64 + fn * 16 + l16) * LDSTR], half);
        for (int fr = 0; fr < 4; ++fr)
            for (int fn = 0; fn < 4; ++fn)
                acc[fr][fn] = __builtin_amdgcn_wmma_f32_16x16x32_bf16(
                    false, af[fr], false, bf[fn], (short)0, acc[fr][fn], false, false);
    }

    // Epilogue: + bias, convert bf16, scatter into Q / K / V^T layouts.
    // Within a block all rows share the same batch index b (128 | 2048),
    // so each fn column resolves to one base pointer + one t-stride.
    const int b = mBlk >> 11;
    for (int fn = 0; fn < 4; ++fn) {
        int nCol = nBlk + waveN * 64 + fn * 16 + l16;
        float bv = bias[nCol];
        int sec = nCol >> 10;               // 0=Q 1=K 2=V
        int c   = nCol & (CDIM - 1);
        int h   = c >> 6;
        int d   = c & (HDIM - 1);
        size_t bh = (size_t)b * NHEAD + h;
        unsigned short* p;
        int strideT;
        if (sec == 0)      { p = Qb + (bh * TLEN) * HDIM + d; strideT = HDIM; }
        else if (sec == 1) { p = Kb + (bh * TLEN) * HDIM + d; strideT = HDIM; }
        else               { p = Vt + (bh * HDIM + d) * TLEN; strideT = 1;    }
        for (int fr = 0; fr < 4; ++fr) {
            int t0 = (mBlk & (TLEN - 1)) + waveM * 64 + fr * 16 + 8 * half;
            for (int r = 0; r < 8; ++r)
                p[(size_t)(t0 + r) * strideT] = f2bf(acc[fr][fn][r] + bv);
        }
    }
}

// ---------------------------------------------------------------------------
// Kernel 2: causal flash attention. One (b,h, 64-query) tile per 128-thread
// block; each wave owns 16 query rows. Key blocks of 32, online softmax.
// ---------------------------------------------------------------------------
#define KSTR 72    // K tile row stride (halfs): 144B, 16B aligned, bank-clean

__global__ __launch_bounds__(128, 1) void flash_attn(
    const unsigned short* __restrict__ Qb, const unsigned short* __restrict__ Kb,
    const unsigned short* __restrict__ Vt, float* __restrict__ out)
{
    __shared__ __align__(16) unsigned short Ks[32 * KSTR];
    __shared__ __align__(16) unsigned short Vs[64 * LDSTR];
    __shared__ __align__(16) unsigned short Ps[4][16 * LDSTR];

    const int tid  = threadIdx.x;
    const int lane = tid & 31;
    const int wave = tid >> 5;
    const int half = lane >> 4;
    const int l16  = lane & 15;

    const int qBase = blockIdx.x * 64;    // grid.x = 32
    const int bh    = blockIdx.y;         // grid.y = 64
    const unsigned short* Qp = Qb + (size_t)bh * TLEN * HDIM;
    const unsigned short* Kp = Kb + (size_t)bh * TLEN * HDIM;
    const unsigned short* Vp = Vt + (size_t)bh * HDIM * TLEN;
    const int q0 = qBase + wave * 16;

    // Resident Q fragments: two A-frags covering hd = 0..31, 32..63.
    v16bf qa[2];
    for (int kk = 0; kk < 2; ++kk) {
        FragBF f;
        const unsigned short* rp = Qp + (size_t)(q0 + l16) * HDIM + kk * 32;
        f.q[0] = *(const uint4*)(rp + 8 * half);
        f.q[1] = *(const uint4*)(rp + 16 + 8 * half);
        qa[kk] = f.v;
    }

    v8f o[4];
    for (int i = 0; i < 4; ++i) o[i] = 0.0f;
    float mrow[8], lrow[8];
    for (int r = 0; r < 8; ++r) { mrow[r] = -1e30f; lrow[r] = 0.0f; }

    const int nkb = (qBase >> 5) + 2;     // causal: keys <= qBase+63
    for (int kb = 0; kb < nkb; ++kb) {
        __syncthreads();
        // Stage K tile (32 keys x 64 hd) and V^T tile (64 hd x 32 keys).
        for (int c = 0; c < 2; ++c) {
            int idx  = tid + 128 * c;
            int row  = idx >> 3;
            int col8 = (idx & 7) * 8;
            *(uint4*)&Ks[row * KSTR + col8] =
                *(const uint4*)(Kp + (size_t)(kb * 32 + row) * HDIM + col8);
            int vrow  = idx >> 2;
            int vcol8 = (idx & 3) * 8;
            *(uint4*)&Vs[vrow * LDSTR + vcol8] =
                *(const uint4*)(Vp + (size_t)vrow * TLEN + kb * 32 + vcol8);
        }
        if (kb + 1 < nkb) {
            __builtin_prefetch(Kp + (size_t)(kb + 1) * 32 * HDIM, 0, 3);
            __builtin_prefetch(Vp + (kb + 1) * 32, 0, 3);
        }
        __syncthreads();

        // Waves whose 16 query rows lie entirely below this key block have
        // nothing unmasked to do; they only keep participating in barriers.
        if (kb * 32 > q0 + 15) {
            __syncthreads();
            continue;
        }

        // S = Q . K^T  (two 16-key column groups, reduce over hd=64).
        v8f s[2];
        for (int j = 0; j < 2; ++j) {
            s[j] = 0.0f;
            for (int kk = 0; kk < 2; ++kk) {
                FragBF f;
                const unsigned short* rp = &Ks[(16 * j + l16) * KSTR + kk * 32];
                f.q[0] = *(const uint4*)(rp + 8 * half);
                f.q[1] = *(const uint4*)(rp + 16 + 8 * half);
                s[j] = __builtin_amdgcn_wmma_f32_16x16x32_bf16(
                    false, qa[kk], false, f.v, (short)0, s[j], false, false);
            }
        }

        // Online softmax (C-layout: lane holds col n=l16, rows r+8*half).
        const int key0 = kb * 32 + l16;
        const int key1 = key0 + 16;
        for (int r = 0; r < 8; ++r) {
            int qrow = q0 + r + 8 * half;
            float v0 = (key0 <= qrow) ? s[0][r] * 0.125f : -1e30f;
            float v1 = (key1 <= qrow) ? s[1][r] * 0.125f : -1e30f;
            float mx = fmaxf(v0, v1);
            mx = fmaxf(mx, __shfl_xor(mx, 1, 32));
            mx = fmaxf(mx, __shfl_xor(mx, 2, 32));
            mx = fmaxf(mx, __shfl_xor(mx, 4, 32));
            mx = fmaxf(mx, __shfl_xor(mx, 8, 32));
            float mnew = fmaxf(mrow[r], mx);
            float al = exp2f((mrow[r] - mnew) * LOG2E);
            float e0 = exp2f((v0 - mnew) * LOG2E);
            float e1 = exp2f((v1 - mnew) * LOG2E);
            float ps = e0 + e1;
            ps += __shfl_xor(ps, 1, 32);
            ps += __shfl_xor(ps, 2, 32);
            ps += __shfl_xor(ps, 4, 32);
            ps += __shfl_xor(ps, 8, 32);
            lrow[r] = lrow[r] * al + ps;
            mrow[r] = mnew;
            for (int no = 0; no < 4; ++no) o[no][r] *= al;
            // C-layout -> A-layout redistribution through per-wave LDS.
            Ps[wave][(r + 8 * half) * LDSTR + l16]      = f2bf(e0);
            Ps[wave][(r + 8 * half) * LDSTR + 16 + l16] = f2bf(e1);
        }
        __syncthreads();

        // O += P . V   (A = P 16x32, B = V^T rows of hd).
        FragBF pf;
        const unsigned short* pr = &Ps[wave][l16 * LDSTR];
        pf.q[0] = *(const uint4*)(pr + 8 * half);
        pf.q[1] = *(const uint4*)(pr + 16 + 8 * half);
        for (int no = 0; no < 4; ++no) {
            FragBF vf;
            const unsigned short* vr = &Vs[(16 * no + l16) * LDSTR];
            vf.q[0] = *(const uint4*)(vr + 8 * half);
            vf.q[1] = *(const uint4*)(vr + 16 + 8 * half);
            o[no] = __builtin_amdgcn_wmma_f32_16x16x32_bf16(
                false, pf.v, false, vf.v, (short)0, o[no], false, false);
        }
    }

    // Normalize and write fp32 output [B, T, C].
    const int b = bh >> 4;
    const int h = bh & (NHEAD - 1);
    for (int no = 0; no < 4; ++no) {
        int d = 16 * no + l16;
        for (int r = 0; r < 8; ++r) {
            int t = q0 + r + 8 * half;
            out[((size_t)b * TLEN + t) * CDIM + h * HDIM + d] = o[no][r] / lrow[r];
        }
    }
}

// ---------------------------------------------------------------------------
extern "C" void kernel_launch(void* const* d_in, const int* in_sizes, int n_in,
                              void* d_out, int out_size, void* d_ws, size_t ws_size,
                              hipStream_t stream) {
    const float* x    = (const float*)d_in[0];   // [4,2048,1024]
    const float* W    = (const float*)d_in[1];   // [3072,1024]
    const float* bias = (const float*)d_in[2];   // [3072]
    float* out = (float*)d_out;                  // [4,2048,1024]

    const size_t qkvElems = (size_t)BATCH * NHEAD * TLEN * HDIM;   // 8.4M halfs each
    unsigned short* Qb = (unsigned short*)d_ws;
    unsigned short* Kb = Qb + qkvElems;
    unsigned short* Vt = Kb + qkvElems;

    qkv_gemm<<<dim3(64, 24), 128, 0, stream>>>(x, W, bias, Qb, Kb, Vt);
    flash_attn<<<dim3(32, 64), 128, 0, stream>>>(Qb, Kb, Vt, out);
}
